// PartBlock_49503793054473
// MI455X (gfx1250) — matrix operand order repaired and checked by reading, not compiled, not run on hardware
//
#include <hip/hip_runtime.h>

typedef unsigned short u16;
typedef unsigned int   u32;
typedef __bf16 bf16_t;
typedef bf16_t v16bf __attribute__((ext_vector_type(16)));
typedef u16    v16u  __attribute__((ext_vector_type(16)));
typedef float  v8f   __attribute__((ext_vector_type(8)));
typedef u32 u32x4 __attribute__((ext_vector_type(4)));
typedef int i32x4 __attribute__((ext_vector_type(4)));
typedef int i32x8 __attribute__((ext_vector_type(8)));

#define NB   32
#define CC   128
#define TT   400
#define VV   27
#define TV   10800
#define SS   2
#define ICN  32
#define RCN  32
#define PTG  75      // 675 p-tiles = 75 groups of 9 (one per wave, 9 waves/block)
#define RSQ  0.9999950000374997f   // 1/sqrt(1+1e-5)

__device__ __forceinline__ float bf2f(u16 h) {
  unsigned int u = ((unsigned int)h) << 16;
  float f; __builtin_memcpy(&f, &u, 4); return f;
}
__device__ __forceinline__ u16 f2bf(float f) {
  unsigned int u; __builtin_memcpy(&u, &f, 4);
  u += 0x7FFFu + ((u >> 16) & 1u);
  return (u16)(u >> 16);
}
__device__ __forceinline__ float lrelu(float v) { return v > 0.f ? v : 0.1f * v; }

// ---- Tensor Data Mover: stage a 16 x Kdim bf16 weight tile (row stride ldw) into LDS ----
// D# group0: count=1, lds_addr, global_addr, type=2. group1: data_size=2B, 2D tensor
// (dim0=Kdim, dim1=16 rows, dim0_stride=ldw), tile 16 x Kdim. Issued by wave 0 only;
// completion via TENSORcnt, published to the block with a barrier.
__device__ __forceinline__ void tdm_stage_w(const u16* __restrict__ Wtile, int ldw,
                                            int Kdim, u16* sW) {
  if (threadIdx.x < 32) {
    unsigned long long ga = (unsigned long long)(size_t)Wtile;
    u32 lds = (u32)(size_t)sW;   // generic LDS addr: low 32 bits = LDS offset
    u32x4 g0 = { 1u, lds, (u32)ga,
                 ((u32)(ga >> 32) & 0x01FFFFFFu) | (2u << 30) };
    i32x8 g1 = { (int)0x00010000u,            // workgroup_mask=0, data_size=1 (2B)
                 (int)((u32)Kdim << 16),      // tensor_dim0[15:0] in bits 63:48
                 (int)(16u << 16),            // tensor_dim0 hi=0, tensor_dim1=16
                 (int)((u32)Kdim << 16),      // tensor_dim1 hi=0, tile_dim0=Kdim
                 16,                          // tile_dim1=16, tile_dim2=0
                 ldw,                         // tensor_dim0_stride lo
                 0, 0 };
    i32x4 z4 = {0, 0, 0, 0};
#if __has_include(<hip/amd_detail/amd_gfx1250_TDM.h>)
    i32x8 z8 = {0, 0, 0, 0, 0, 0, 0, 0};
    __builtin_amdgcn_tensor_load_to_lds(g0, g1, z4, z4, z8, 0);
#else
    __builtin_amdgcn_tensor_load_to_lds(g0, g1, z4, z4, 0);
#endif
    __builtin_amdgcn_s_wait_tensorcnt(0);
  }
  __syncthreads();
}

// One 16x16 wave tile: D = sA(16 x Kdim, LDS) * X[0:K, p0:p0+16] (bf16, row stride ldx).
__device__ __forceinline__ v8f wmma_loop_lds(const u16* sA, int Kdim,
                                             const u16* __restrict__ X, int ldx, int p0) {
  const int lane = threadIdx.x & 31;
  const int hi   = lane >> 4;
  const int acol = hi * 8;      // A: lanes 16-31 hold K+8 within each 16-block
  const int bko  = hi * 8;      // B: lanes 16-31 hold K+8 within each 16-block
  const u16* arow = sA + (lane & 15) * Kdim;
  const u16* xcol = X + (p0 + (lane & 15));
  v8f acc = {0.f,0.f,0.f,0.f,0.f,0.f,0.f,0.f};
  for (int k0 = 0; k0 < Kdim; k0 += 32) {
    v16u au, bu;
#pragma unroll
    for (int j = 0; j < 8; ++j) {
      int kk = k0 + ((j & 4) << 2) + acol + ((j & 3) << 1);
      au[2*j]   = arow[kk];
      au[2*j+1] = arow[kk+1];
    }
#pragma unroll
    for (int i = 0; i < 16; ++i) bu[i] = xcol[(long)(k0 + ((i & 8) << 1) + bko + (i & 7)) * ldx];
    if (k0 + 32 < Kdim) __builtin_prefetch(xcol + (long)(k0 + 32 + bko) * ldx, 0, 0);
    acc = __builtin_amdgcn_wmma_f32_16x16x32_bf16(
        false, __builtin_bit_cast(v16bf, au),
        false, __builtin_bit_cast(v16bf, bu),
        (short)0, acc, false, false);
  }
  return acc;
}

// ---------------- prep kernels ----------------

__global__ void k_pe(float* pe) {  // pe[c*27+v]
  int idx = blockIdx.x * blockDim.x + threadIdx.x;
  if (idx >= CC * VV) return;
  int c = idx / VV, v = idx % VV;
  float ceven = (float)(c & ~1);
  float div = expf(ceven * (-9.210340371976184f / (float)CC));
  float ang = (float)v * div;
  pe[idx] = (c & 1) ? cosf(ang) : sinf(ang);
}

__global__ void k_cvt(const float* src, u16* dst, int n) {
  int i = blockIdx.x * blockDim.x + threadIdx.x;
  if (i < n) dst[i] = f2bf(src[i]);
}

__global__ void k_wcat(const float* back_w, const float* back_bn, const float* pw_w,
                       const float* bn_b, u16* wcatb, float* cbias) {
  int idx = blockIdx.x * blockDim.x + threadIdx.x;
  if (idx >= CC * 96) return;
  int o = idx / 96, q = idx % 96, i = q / 32, r = q % 32;
  float bsc = back_bn[o] * RSQ;
  float s = 0.f;
#pragma unroll 8
  for (int m = 0; m < 32; ++m)
    s += back_w[o*32 + m] * (bn_b[i*64 + m] * RSQ) * pw_w[i*1024 + m*32 + r];
  wcatb[o*96 + q] = f2bf(bsc * s);
  if (q == 0) {
    float sb = 0.f;
    for (int ii = 0; ii < 3; ++ii)
      for (int m = 0; m < 32; ++m)
        sb += back_w[o*32 + m] * bn_b[ii*64 + 32 + m];
    cbias[o] = bsc * sb + back_bn[CC + o];
  }
}

// ---------------- stage 1: qk = in_w @ (x + pe) + in_b ----------------

__global__ void k_qk(const u16* __restrict__ Wb, const float* __restrict__ in_b,
                     const float* __restrict__ x, const float* __restrict__ pe,
                     u16* __restrict__ qkbuf) {
  __shared__ u16 sW[16 * CC];
  const int mt_n = CC / 16;
  int bid = blockIdx.x;
  int n   = bid / (mt_n * PTG);
  int rem = bid % (mt_n * PTG);
  int m0  = (rem / PTG) * 16;
  int ptg = rem % PTG;
  tdm_stage_w(Wb + (size_t)m0 * CC, CC, CC, sW);

  int w = threadIdx.x >> 5;
  int p0 = (ptg * 9 + w) * 16;
  const int lane = threadIdx.x & 31;
  const int hi = lane >> 4, acol = hi * 8, bko = hi * 8;
  const u16* arow = sW + (lane & 15) * CC;
  int p = p0 + (lane & 15);
  int v = p % VV;
  const float* xb = x + (long)n * CC * TV;
  v8f acc = {0.f,0.f,0.f,0.f,0.f,0.f,0.f,0.f};
  for (int k0 = 0; k0 < CC; k0 += 32) {
    v16u au, bu;
#pragma unroll
    for (int j = 0; j < 8; ++j) {
      int kk = k0 + ((j & 4) << 2) + acol + ((j & 3) << 1);
      au[2*j]   = arow[kk];
      au[2*j+1] = arow[kk+1];
    }
#pragma unroll
    for (int i = 0; i < 16; ++i) {
      int k = k0 + ((i & 8) << 1) + bko + (i & 7);
      bu[i] = f2bf(xb[(long)k * TV + p] + pe[k * VV + v]);
    }
    acc = __builtin_amdgcn_wmma_f32_16x16x32_bf16(
        false, __builtin_bit_cast(v16bf, au), false, __builtin_bit_cast(v16bf, bu),
        (short)0, acc, false, false);
  }
#pragma unroll
  for (int j = 0; j < 8; ++j) {
    int o = m0 + j + 8 * hi;
    qkbuf[((long)n * CC + o) * TV + p] = f2bf(acc[j] + in_b[o]);
  }
}

// ------- stage 2: att[n,s,u,v] = tanh(sum_{c,t} q*k / (IC*T)) * alpha + att0 (padded 32x32) -------

__global__ void k_att(const u16* __restrict__ qkbuf, const float* __restrict__ alphas,
                      const float* __restrict__ att0, u16* __restrict__ attb) {
  int ns = blockIdx.x;                  // 64 blocks, blockDim 128 (4 waves)
  int n = ns >> 1, s = ns & 1;
  int w = threadIdx.x >> 5;
  int mi = w >> 1, ni = w & 1;
  int lane = threadIdx.x & 31;
  int hi = lane >> 4, acol = hi * 8, bko = hi * 8;
  int u = mi * 16 + (lane & 15);
  int vc = ni * 16 + (lane & 15);
  const u16* qb = qkbuf + ((long)n * CC + s * ICN) * TV;
  const u16* kb = qkbuf + ((long)n * CC + 2 * ICN + s * ICN) * TV;
  v8f acc = {0.f,0.f,0.f,0.f,0.f,0.f,0.f,0.f};
  const int Kdim = ICN * TT;  // 12800
  for (int k0 = 0; k0 < Kdim; k0 += 32) {
    v16u au, bu;
#pragma unroll
    for (int j = 0; j < 8; ++j) {
#pragma unroll
      for (int low = 0; low < 2; ++low) {
        int kk = k0 + ((j & 4) << 2) + acol + ((j & 3) << 1) + low;
        int c = kk / TT, t = kk - c * TT;
        au[2*j + low] = (u < VV) ? qb[(long)c * TV + t * VV + u] : (u16)0;
      }
    }
#pragma unroll
    for (int i = 0; i < 16; ++i) {
      int kk = k0 + ((i & 8) << 1) + bko + (i & 7);
      int c = kk / TT, t = kk - c * TT;
      bu[i] = (vc < VV) ? kb[(long)c * TV + t * VV + vc] : (u16)0;
    }
    acc = __builtin_amdgcn_wmma_f32_16x16x32_bf16(
        false, __builtin_bit_cast(v16bf, au), false, __builtin_bit_cast(v16bf, bu),
        (short)0, acc, false, false);
  }
  float inv = 1.0f / (float)(ICN * TT);
  float al = alphas[s];
#pragma unroll
  for (int j = 0; j < 8; ++j) {
    int uu = mi * 16 + j + 8 * hi;
    float val = 0.f;
    if (uu < VV && vc < VV)
      val = tanhf(acc[j] * inv) * al + att0[(s * VV + uu) * VV + vc];
    attb[((long)(n * SS + s) * 32 + uu) * 32 + vc] = f2bf(val);
  }
}

// ------- stage 3: y2[n, s*C+c, t, v] = sum_u x[n,c,t,u] * att[n,s,u,v]  (K=32 padded) -------

__global__ void k_apply(const float* __restrict__ x, const u16* __restrict__ attb,
                        u16* __restrict__ y2b) {
  int wid = blockIdx.x * (blockDim.x >> 5) + (threadIdx.x >> 5);
  const int mt_n = (CC * TT) / 16;   // 3200 row tiles (rows = (c,t))
  const int per_n = mt_n * 2;
  if (wid >= NB * per_n) return;
  int n = wid / per_n, r = wid % per_n;
  int mt = r >> 1, nt = r & 1;
  int lane = threadIdx.x & 31;
  int hi = lane >> 4, acol = hi * 8, bko = hi * 8;
  int m = mt * 16 + (lane & 15);
  int c = m / TT, t = m - c * TT;
  const float* xrow = x + ((long)n * CC + c) * TV + t * VV;
  v16u au;
#pragma unroll
  for (int j = 0; j < 8; ++j) {
#pragma unroll
    for (int low = 0; low < 2; ++low) {
      int u = ((j & 4) << 2) + acol + ((j & 3) << 1) + low;
      au[2*j + low] = (u < VV) ? f2bf(xrow[u]) : (u16)0;
    }
  }
  int vc = nt * 16 + (lane & 15);
  const u16* ab = attb + (long)n * SS * 1024;
  v16u b0, b1;
#pragma unroll
  for (int i = 0; i < 16; ++i) {
    int u = ((i & 8) << 1) + bko + (i & 7);
    b0[i] = ab[u * 32 + vc];
    b1[i] = ab[1024 + u * 32 + vc];
  }
  v8f z = {0.f,0.f,0.f,0.f,0.f,0.f,0.f,0.f};
  v8f acc0 = __builtin_amdgcn_wmma_f32_16x16x32_bf16(
      false, __builtin_bit_cast(v16bf, au), false, __builtin_bit_cast(v16bf, b0),
      (short)0, z, false, false);
  v8f acc1 = __builtin_amdgcn_wmma_f32_16x16x32_bf16(
      false, __builtin_bit_cast(v16bf, au), false, __builtin_bit_cast(v16bf, b1),
      (short)0, z, false, false);
  if (vc < VV) {
#pragma unroll
    for (int j = 0; j < 8; ++j) {
      int pd = mt * 16 + j + 8 * hi;
      int cc = pd / TT, tt = pd - cc * TT;
      long base = (long)n * (SS * CC) * TV;
      y2b[base + ((long)cc) * TV + tt * VV + vc] = f2bf(acc0[j]);
      y2b[base + ((long)(CC + cc)) * TV + tt * VV + vc] = f2bf(acc1[j]);
    }
  }
}

// ------- stage 4: y3 = lrelu(x + bn(out_w @ y2 + out_b)) -------

__global__ void k_out(const u16* __restrict__ Wb, const float* __restrict__ out_b,
                      const float* __restrict__ out_bn, const float* __restrict__ x,
                      const u16* __restrict__ y2b, u16* __restrict__ y3b) {
  __shared__ u16 sW[16 * SS * CC];
  const int mt_n = CC / 16;
  int bid = blockIdx.x;
  int n   = bid / (mt_n * PTG);
  int rem = bid % (mt_n * PTG);
  int m0  = (rem / PTG) * 16;
  int ptg = rem % PTG;
  tdm_stage_w(Wb + (size_t)m0 * (SS * CC), SS * CC, SS * CC, sW);
  int p0 = (ptg * 9 + (threadIdx.x >> 5)) * 16;
  v8f d = wmma_loop_lds(sW, SS * CC, y2b + (long)n * SS * CC * TV, TV, p0);
  int lane = threadIdx.x & 31;
  int col = p0 + (lane & 15), hi = lane >> 4;
#pragma unroll
  for (int j = 0; j < 8; ++j) {
    int o = m0 + j + 8 * hi;
    float val = (d[j] + out_b[o]) * (out_bn[o] * RSQ) + out_bn[CC + o];
    val = lrelu(x[((long)n * CC + o) * TV + col] + val);
    y3b[((long)n * CC + o) * TV + col] = f2bf(val);
  }
}

// ------- stage 5: y4 = lrelu(x + bn(ff_w @ y3 + ff_b));  store f32 res + bf16 -------

__global__ void k_ff(const u16* __restrict__ Wb, const float* __restrict__ ff_b,
                     const float* __restrict__ ff_bn, const float* __restrict__ x,
                     const u16* __restrict__ y3b, float* __restrict__ resf,
                     u16* __restrict__ y4b) {
  __shared__ u16 sW[16 * CC];
  const int mt_n = CC / 16;
  int bid = blockIdx.x;
  int n   = bid / (mt_n * PTG);
  int rem = bid % (mt_n * PTG);
  int m0  = (rem / PTG) * 16;
  int ptg = rem % PTG;
  tdm_stage_w(Wb + (size_t)m0 * CC, CC, CC, sW);
  int p0 = (ptg * 9 + (threadIdx.x >> 5)) * 16;
  v8f d = wmma_loop_lds(sW, CC, y3b + (long)n * CC * TV, TV, p0);
  int lane = threadIdx.x & 31;
  int col = p0 + (lane & 15), hi = lane >> 4;
#pragma unroll
  for (int j = 0; j < 8; ++j) {
    int o = m0 + j + 8 * hi;
    float val = (d[j] + ff_b[o]) * (ff_bn[o] * RSQ) + ff_bn[CC + o];
    val = lrelu(x[((long)n * CC + o) * TV + col] + val);
    long idx = ((long)n * CC + o) * TV + col;
    resf[idx] = val;
    y4b[idx] = f2bf(val);
  }
}

// ------- stage 6: yd = lrelu(bn(down_w @ y4)) -------

__global__ void k_down(const u16* __restrict__ Wb, const float* __restrict__ down_bn,
                       const u16* __restrict__ y4b, u16* __restrict__ ydb) {
  __shared__ u16 sW[16 * CC];
  const int mt_n = RCN / 16;
  int bid = blockIdx.x;
  int n   = bid / (mt_n * PTG);
  int rem = bid % (mt_n * PTG);
  int m0  = (rem / PTG) * 16;
  int ptg = rem % PTG;
  tdm_stage_w(Wb + (size_t)m0 * CC, CC, CC, sW);
  int p0 = (ptg * 9 + (threadIdx.x >> 5)) * 16;
  v8f d = wmma_loop_lds(sW, CC, y4b + (long)n * CC * TV, TV, p0);
  int lane = threadIdx.x & 31;
  int col = p0 + (lane & 15), hi = lane >> 4;
#pragma unroll
  for (int j = 0; j < 8; ++j) {
    int o = m0 + j + 8 * hi;
    float val = d[j] * (down_bn[o] * RSQ) + down_bn[RCN + o];
    ydb[((long)n * RCN + o) * TV + col] = f2bf(lrelu(val));
  }
}

// ------- stage 7: three dilated depthwise temporal convs -> hcat (n, 96, t, v) -------

__global__ void k_dw(const u16* __restrict__ ydb, const float* __restrict__ dw_w,
                     const float* __restrict__ dw_b, const float* __restrict__ bn_a,
                     u16* __restrict__ hcat) {
  int idx = blockIdx.x * blockDim.x + threadIdx.x;
  const int total = NB * 96 * TV;
  if (idx >= total) return;
  int p = idx % TV;
  int q = idx / TV;
  int ch = q % 96, n = q / 96;
  int i = ch / RCN, r = ch % RCN, dil = i + 1;
  int t = p / VV, v = p - t * VV;
  const u16* src = ydb + ((long)n * RCN + r) * TV;
  float w0 = dw_w[(i * RCN + r) * 3 + 0];
  float w1 = dw_w[(i * RCN + r) * 3 + 1];
  float w2 = dw_w[(i * RCN + r) * 3 + 2];
  float acc = dw_b[i * RCN + r];
  int tm = t - dil, tp = t + dil;
  if (tm >= 0) acc += w0 * bf2f(src[tm * VV + v]);
  acc += w1 * bf2f(src[t * VV + v]);
  if (tp < TT) acc += w2 * bf2f(src[tp * VV + v]);
  acc = acc * (bn_a[i * 64 + r] * RSQ) + bn_a[i * 64 + RCN + r];
  hcat[((long)n * 96 + ch) * TV + p] = f2bf(lrelu(acc));
}

// ------- stage 8: out = lrelu(Wcat @ hcat + cbias + res) -------

__global__ void k_final(const u16* __restrict__ Wb, const float* __restrict__ cbias,
                        const float* __restrict__ resf, const u16* __restrict__ hcat,
                        float* __restrict__ outp) {
  __shared__ u16 sW[16 * 96];
  const int mt_n = CC / 16;
  int bid = blockIdx.x;
  int n   = bid / (mt_n * PTG);
  int rem = bid % (mt_n * PTG);
  int m0  = (rem / PTG) * 16;
  int ptg = rem % PTG;
  tdm_stage_w(Wb + (size_t)m0 * 96, 96, 96, sW);
  int p0 = (ptg * 9 + (threadIdx.x >> 5)) * 16;
  v8f d = wmma_loop_lds(sW, 96, hcat + (long)n * 96 * TV, TV, p0);
  int lane = threadIdx.x & 31;
  int col = p0 + (lane & 15), hi = lane >> 4;
#pragma unroll
  for (int j = 0; j < 8; ++j) {
    int o = m0 + j + 8 * hi;
    long idx = ((long)n * CC + o) * TV + col;
    outp[idx] = lrelu(d[j] + cbias[o] + resf[idx]);
  }
}

// ---------------- host launcher ----------------

extern "C" void kernel_launch(void* const* d_in, const int* in_sizes, int n_in,
                              void* d_out, int out_size, void* d_ws, size_t ws_size,
                              hipStream_t stream) {
  (void)in_sizes; (void)n_in; (void)out_size; (void)ws_size;
  const float* x       = (const float*)d_in[0];
  const float* in_w    = (const float*)d_in[1];
  const float* in_b    = (const float*)d_in[2];
  const float* alphas  = (const float*)d_in[3];
  const float* att0    = (const float*)d_in[4];
  const float* out_w   = (const float*)d_in[5];
  const float* out_b   = (const float*)d_in[6];
  const float* out_bn  = (const float*)d_in[7];
  const float* ff_w    = (const float*)d_in[8];
  const float* ff_b    = (const float*)d_in[9];
  const float* ff_bn   = (const float*)d_in[10];
  const float* down_w  = (const float*)d_in[11];
  const float* down_bn = (const float*)d_in[12];
  const float* dw_w    = (const float*)d_in[13];
  const float* dw_b    = (const float*)d_in[14];
  const float* bn_a    = (const float*)d_in[15];
  const float* pw_w    = (const float*)d_in[16];
  const float* bn_b    = (const float*)d_in[17];
  const float* back_w  = (const float*)d_in[18];
  const float* back_bn = (const float*)d_in[19];

  char* ws = (char*)d_ws;
  size_t off = 0;
  auto alloc = [&](size_t bytes) -> char* {
    char* p = ws + off;
    off = (off + bytes + 255) & ~(size_t)255;
    return p;
  };
  float* pe     = (float*)alloc((size_t)CC * VV * 4);
  u16*  in_wb   = (u16*)alloc((size_t)CC * CC * 2);
  u16*  out_wb  = (u16*)alloc((size_t)CC * SS * CC * 2);
  u16*  ff_wb   = (u16*)alloc((size_t)CC * CC * 2);
  u16*  down_wb = (u16*)alloc((size_t)RCN * CC * 2);
  u16*  wcatb   = (u16*)alloc((size_t)CC * 96 * 2);
  float* cbias  = (float*)alloc((size_t)CC * 4);
  u16*  attb    = (u16*)alloc((size_t)NB * SS * 32 * 32 * 2);
  // big aliased regions
  u16*  regA    = (u16*)alloc((size_t)NB * CC * TV * 2);       // qk -> y3b -> hcat
  char* regB    = alloc((size_t)NB * SS * CC * TV * 2);        // y2b (bf16) -> res (f32)
  u16*  y4b     = (u16*)alloc((size_t)NB * CC * TV * 2);
  u16*  ydb     = (u16*)alloc((size_t)NB * RCN * TV * 2);

  // prep
  k_pe<<<(CC * VV + 255) / 256, 256, 0, stream>>>(pe);
  k_cvt<<<(CC * CC + 255) / 256, 256, 0, stream>>>(in_w, in_wb, CC * CC);
  k_cvt<<<(CC * SS * CC + 255) / 256, 256, 0, stream>>>(out_w, out_wb, CC * SS * CC);
  k_cvt<<<(CC * CC + 255) / 256, 256, 0, stream>>>(ff_w, ff_wb, CC * CC);
  k_cvt<<<(RCN * CC + 255) / 256, 256, 0, stream>>>(down_w, down_wb, RCN * CC);
  k_wcat<<<(CC * 96 + 255) / 256, 256, 0, stream>>>(back_w, back_bn, pw_w, bn_b, wcatb, cbias);

  const int gemm_blocks = NB * (CC / 16) * PTG;   // 19200 blocks, 9 waves each
  const int down_blocks = NB * (RCN / 16) * PTG;  // 4800

  k_qk<<<gemm_blocks, 288, 0, stream>>>(in_wb, in_b, x, pe, regA);
  k_att<<<NB * SS, 128, 0, stream>>>(regA, alphas, att0, attb);
  k_apply<<<(NB * (CC * TT / 16) * 2) / 8, 256, 0, stream>>>(x, attb, (u16*)regB);
  k_out<<<gemm_blocks, 288, 0, stream>>>(out_wb, out_b, out_bn, x, (u16*)regB, regA);
  k_ff<<<gemm_blocks, 288, 0, stream>>>(ff_wb, ff_b, ff_bn, x, regA, (float*)regB, y4b);
  k_down<<<down_blocks, 288, 0, stream>>>(down_wb, down_bn, y4b, ydb);
  k_dw<<<(NB * 96 * TV + 255) / 256, 256, 0, stream>>>(ydb, dw_w, dw_b, bn_a, regA);
  k_final<<<gemm_blocks, 288, 0, stream>>>(wcatb, cbias, (float*)regB, regA, (float*)d_out);
}
